// ObservationEmbeddingV2_6219112644929
// MI455X (gfx1250) — matrix operand order repaired and checked by reading, not compiled, not run on hardware
//
#include <hip/hip_runtime.h>
#include <hip/hip_bf16.h>

typedef __bf16 bf16;
typedef __attribute__((ext_vector_type(16))) __bf16 v16bf;
typedef __attribute__((ext_vector_type(8)))  __bf16 v8bf;
typedef __attribute__((ext_vector_type(8)))  float  v8f;

#define B_TOTAL 16384
#define D_MODEL 1024
#define HIDDEN  2048
#define G0_IN   512
#define REST_IN 256
#define SEQ_LEN 7
#define OBS_W   2048
#define M_TILE  32            // batch rows per workgroup
#define HC      256           // hidden chunk per iteration (8 waves x 32 cols)
#define XS_STRIDE (G0_IN + 8) // padded LDS row strides (16B-aligned rows)
#define HS_STRIDE (HC + 8)

// Build a 16-element bf16 A/B fragment from two 16-byte LDS loads
__device__ __forceinline__ v16bf ld_frag(const bf16* p0, const bf16* p1) {
  v8bf lo = *(const v8bf*)p0;
  v8bf hi = *(const v8bf*)p1;
  return __builtin_shufflevector(lo, hi, 0,1,2,3,4,5,6,7,8,9,10,11,12,13,14,15);
}

__global__ void cvt_f32_to_bf16(const float* __restrict__ src, bf16* __restrict__ dst, int n) {
  int i = blockIdx.x * 256 + threadIdx.x;
  int stride = gridDim.x * 256;
  for (; i < n; i += stride) dst[i] = (bf16)src[i];
}

__global__ __launch_bounds__(256) void fused_groupmlp(
    const float* __restrict__ obs,
    const bf16* __restrict__ w1g0, const bf16* __restrict__ w3g0, const bf16* __restrict__ w2g0,
    const bf16* __restrict__ w1r,  const bf16* __restrict__ w3r,  const bf16* __restrict__ w2r,
    const float* __restrict__ norm_w, const float* __restrict__ norm_b,
    const float* __restrict__ term_embed,
    float* __restrict__ out)
{
  __shared__ __attribute__((aligned(16))) bf16 xs[M_TILE][XS_STRIDE];
  __shared__ __attribute__((aligned(16))) bf16 hs[M_TILE][HS_STRIDE];
  __shared__ float rowsq[M_TILE];

  const int s    = blockIdx.y;                 // sequence/group index 0..6
  const int b0   = blockIdx.x * M_TILE;        // batch row base
  const int tid  = threadIdx.x;
  const int lane = tid & 31;
  const int wv   = tid >> 5;                   // wave id 0..7
  const int half = lane >> 4;                  // 0|1 (wave32 half)
  const int l16  = lane & 15;

  const bf16 *w1p, *w3p, *w2p;
  int K1, xoff, kshift;
  if (s == 0) {
    w1p = w1g0; w3p = w3g0; w2p = w2g0;
    K1 = G0_IN; xoff = 0; kshift = 9;
  } else {
    int g = s - 1;
    w1p = w1r + (size_t)g * HIDDEN * REST_IN;
    w3p = w3r + (size_t)g * HIDDEN * REST_IN;
    w2p = w2r + (size_t)g * D_MODEL * HIDDEN;
    K1 = REST_IN; xoff = G0_IN + g * REST_IN; kshift = 8;
  }

  // ---- load x tile (M_TILE x K1) -> LDS as bf16 ----
  const int nx = M_TILE * K1;
  for (int i = tid; i < nx; i += 256) {
    int r = i >> kshift;
    int c = i & (K1 - 1);
    xs[r][c] = (bf16)obs[(size_t)(b0 + r) * OBS_W + xoff + c];
  }
  if (tid < M_TILE) rowsq[tid] = 0.0f;
  __syncthreads();

  const v8f zero8 = {0.f,0.f,0.f,0.f,0.f,0.f,0.f,0.f};
  v8f acc[2][8];
  #pragma unroll
  for (int rt = 0; rt < 2; ++rt)
    #pragma unroll
    for (int nt = 0; nt < 8; ++nt) acc[rt][nt] = zero8;

  #pragma unroll 1
  for (int ch = 0; ch < HIDDEN / HC; ++ch) {
    // ---- stage A: this wave computes gate/val for hidden cols [ch*HC + wv*32, +32) ----
    v8f gacc[2][2], vacc[2][2];
    #pragma unroll
    for (int rt = 0; rt < 2; ++rt)
      #pragma unroll
      for (int ct = 0; ct < 2; ++ct) { gacc[rt][ct] = zero8; vacc[rt][ct] = zero8; }

    #pragma unroll 1
    for (int kk = 0; kk < K1; kk += 32) {
      const int abase = kk + half * 16;
      v16bf afr[2];
      #pragma unroll
      for (int rt = 0; rt < 2; ++rt) {
        const int row = rt * 16 + l16;
        afr[rt] = ld_frag(&xs[row][abase], &xs[row][abase + 8]);
      }
      #pragma unroll
      for (int ct = 0; ct < 2; ++ct) {
        const int n = ch * HC + wv * 32 + ct * 16 + l16;        // hidden index = weight row
        const bf16* bp1 = w1p + (size_t)n * K1 + kk + half * 16; // contiguous 32B per lane
        const bf16* bp3 = w3p + (size_t)n * K1 + kk + half * 16;
        v16bf b1 = *(const v16bf*)bp1;
        v16bf b3 = *(const v16bf*)bp3;
        #pragma unroll
        for (int rt = 0; rt < 2; ++rt) {
          gacc[rt][ct] = __builtin_amdgcn_wmma_f32_16x16x32_bf16(
              false, afr[rt], false, b1, (short)0, gacc[rt][ct], false, false);
          vacc[rt][ct] = __builtin_amdgcn_wmma_f32_16x16x32_bf16(
              false, afr[rt], false, b3, (short)0, vacc[rt][ct], false, false);
        }
      }
    }

    // silu(gate) * val -> bf16 h chunk in LDS (D-fragment layout -> row major)
    // silu(g) = g * sigmoid(g); use fast v_rcp_f32 instead of IEEE divide
    #pragma unroll
    for (int rt = 0; rt < 2; ++rt)
      #pragma unroll
      for (int ct = 0; ct < 2; ++ct) {
        const int col = wv * 32 + ct * 16 + l16;
        #pragma unroll
        for (int r = 0; r < 8; ++r) {
          float g = gacc[rt][ct][r];
          float v = vacc[rt][ct][r];
          float sig = __builtin_amdgcn_rcpf(1.0f + __expf(-g));
          float h = g * sig * v;
          hs[rt * 16 + r + 8 * half][col] = (bf16)h;
        }
      }
    __syncthreads();

    // ---- stage B: acc += h_chunk @ W2^T for this wave's 128 output cols ----
    #pragma unroll 1
    for (int ks = 0; ks < HC / 32; ++ks) {
      const int abase = ks * 32 + half * 16;
      v16bf afr[2];
      #pragma unroll
      for (int rt = 0; rt < 2; ++rt) {
        const int row = rt * 16 + l16;
        afr[rt] = ld_frag(&hs[row][abase], &hs[row][abase + 8]);
      }
      #pragma unroll
      for (int nt = 0; nt < 8; ++nt) {
        const int n = wv * 128 + nt * 16 + l16;                 // output col = w2 row
        const bf16* bp = w2p + (size_t)n * HIDDEN + ch * HC + ks * 32 + half * 16;
        __builtin_prefetch((const void*)(bp + 32), 0, 0);       // next k-slice
        v16bf bfr = *(const v16bf*)bp;
        #pragma unroll
        for (int rt = 0; rt < 2; ++rt)
          acc[rt][nt] = __builtin_amdgcn_wmma_f32_16x16x32_bf16(
              false, afr[rt], false, bfr, (short)0, acc[rt][nt], false, false);
      }
    }
    __syncthreads();
  }

  // ---- epilogue: RMS norm (mean of squares over 1024), affine, + term_embed ----
  float sq[2][8];
  #pragma unroll
  for (int rt = 0; rt < 2; ++rt)
    #pragma unroll
    for (int r = 0; r < 8; ++r) {
      float a = 0.f;
      #pragma unroll
      for (int nt = 0; nt < 8; ++nt) { float x = acc[rt][nt][r]; a += x * x; }
      a += __shfl_xor(a, 1, 32);
      a += __shfl_xor(a, 2, 32);
      a += __shfl_xor(a, 4, 32);
      a += __shfl_xor(a, 8, 32);   // summed over 16-lane group (16 columns)
      sq[rt][r] = a;
    }
  if (l16 == 0) {
    #pragma unroll
    for (int rt = 0; rt < 2; ++rt)
      #pragma unroll
      for (int r = 0; r < 8; ++r)
        atomicAdd(&rowsq[rt * 16 + r + 8 * half], sq[rt][r]);
  }
  __syncthreads();

  float rinv[2][8];
  #pragma unroll
  for (int rt = 0; rt < 2; ++rt)
    #pragma unroll
    for (int r = 0; r < 8; ++r)
      rinv[rt][r] = __builtin_amdgcn_rsqf(
          rowsq[rt * 16 + r + 8 * half] * (1.0f / (float)D_MODEL) + 1e-5f);

  #pragma unroll
  for (int nt = 0; nt < 8; ++nt) {
    const int col = wv * 128 + nt * 16 + l16;
    const float nw = norm_w[col];
    const float nb = norm_b[col];
    const float te = term_embed[s * D_MODEL + col];
    #pragma unroll
    for (int rt = 0; rt < 2; ++rt)
      #pragma unroll
      for (int r = 0; r < 8; ++r) {
        const int rowl = rt * 16 + r + 8 * half;
        const float v = acc[rt][nt][r] * rinv[rt][r] * nw + nb + te;
        out[(size_t)(b0 + rowl) * (SEQ_LEN * D_MODEL) + (size_t)s * D_MODEL + col] = v;
      }
  }
}

extern "C" void kernel_launch(void* const* d_in, const int* in_sizes, int n_in,
                              void* d_out, int out_size, void* d_ws, size_t ws_size,
                              hipStream_t stream) {
  const float* obs   = (const float*)d_in[0];
  const float* w1g0f = (const float*)d_in[1];
  const float* w3g0f = (const float*)d_in[2];
  const float* w2g0f = (const float*)d_in[3];
  const float* w1rf  = (const float*)d_in[4];
  const float* w3rf  = (const float*)d_in[5];
  const float* w2rf  = (const float*)d_in[6];
  const float* norm_w = (const float*)d_in[7];
  const float* norm_b = (const float*)d_in[8];
  const float* term_e = (const float*)d_in[9];
  float* out = (float*)d_out;

  // bf16 weight staging in workspace (all weights ~44 MB bf16, L2-resident)
  bf16* ws = (bf16*)d_ws;
  const size_t n_w1g0 = (size_t)HIDDEN * G0_IN;        // 1,048,576
  const size_t n_w2g0 = (size_t)D_MODEL * HIDDEN;      // 2,097,152
  const size_t n_w1r  = (size_t)6 * HIDDEN * REST_IN;  // 3,145,728
  const size_t n_w2r  = (size_t)6 * D_MODEL * HIDDEN;  // 12,582,912

  bf16* w1g0 = ws;
  bf16* w3g0 = w1g0 + n_w1g0;
  bf16* w2g0 = w3g0 + n_w1g0;
  bf16* w1r  = w2g0 + n_w2g0;
  bf16* w3r  = w1r  + n_w1r;
  bf16* w2r  = w3r  + n_w1r;

  struct Job { const float* src; bf16* dst; int n; };
  const Job jobs[6] = {
    { w1g0f, w1g0, (int)n_w1g0 },
    { w3g0f, w3g0, (int)n_w1g0 },
    { w2g0f, w2g0, (int)n_w2g0 },
    { w1rf,  w1r,  (int)n_w1r  },
    { w3rf,  w3r,  (int)n_w1r  },
    { w2rf,  w2r,  (int)n_w2r  },
  };
  for (int j = 0; j < 6; ++j) {
    int blocks = (jobs[j].n + 256 * 8 - 1) / (256 * 8);
    if (blocks > 4096) blocks = 4096;
    cvt_f32_to_bf16<<<blocks, 256, 0, stream>>>(jobs[j].src, jobs[j].dst, jobs[j].n);
  }

  dim3 grid(B_TOTAL / M_TILE, SEQ_LEN);  // 512 x 7 workgroups
  fused_groupmlp<<<grid, 256, 0, stream>>>(
      obs, w1g0, w3g0, w2g0, w1r, w3r, w2r, norm_w, norm_b, term_e, out);
}
